// JetEfficiencyNet_6373731468041
// MI455X (gfx1250) — compile-verified, matrix operand order checked by reading, not run on hardware
//
#include <hip/hip_runtime.h>

typedef __attribute__((ext_vector_type(16))) _Float16 v16h;
typedef __attribute__((ext_vector_type(8)))  float    v8f;

#define BB 128
#define NN 64
#define ROWS (BB*NN)      // 8192
#define XSTR 160          // x buffer feature stride (IN_F + 128)
#define HSTR 192          // hi/hj feature stride (max att_mid)
#define LJ   201          // LDS row stride for hj tile (gcd(201,64)=1 -> conflict free)

// Branch-free tanh: stable form, hardware v_exp_f32 + v_rcp_f32. |err| ~1e-6.
__device__ __forceinline__ float fast_tanh(float x) {
    float ax = __builtin_fabsf(x);
    float t  = __expf(-2.f * ax);                       // e^{-2|x|} in (0,1]
    float r  = (1.f - t) * __builtin_amdgcn_rcpf(1.f + t);
    return __builtin_copysignf(r, x);
}

// ---------------------------------------------------------------------------
// x[:,0:32] = inp
__global__ void copy_inp_kernel(const float* __restrict__ inp, float* __restrict__ x) {
    int idx = blockIdx.x * blockDim.x + threadIdx.x;
    if (idx < ROWS * 32) {
        int r = idx >> 5, c = idx & 31;
        x[r * XSTR + c] = inp[idx];
    }
}

// ---------------------------------------------------------------------------
// hi[r,m] = sum_k x[r,k]*W1[m,k] + b1[m] ; hj[r,m] = sum_k x[r,k]*W1[m,F+k]
// blockDim.x == att_mid (64 or 192), grid = ROWS
__global__ void hi_hj_kernel(const float* __restrict__ x, const float* __restrict__ W1,
                             const float* __restrict__ b1,
                             float* __restrict__ hi, float* __restrict__ hj, int F) {
    __shared__ float xr[XSTR];
    const int r = blockIdx.x, m = threadIdx.x, M = blockDim.x;
    for (int k = m; k < F; k += M) xr[k] = x[r * XSTR + k];
    __syncthreads();
    const float* w = W1 + (size_t)m * (2 * F);
    float a = 0.f, b = 0.f;
    for (int k = 0; k < F; ++k) {
        float xv = xr[k];
        a = fmaf(xv, w[k], a);
        b = fmaf(xv, w[F + k], b);
    }
    hi[r * HSTR + m] = a + b1[m];
    hj[r * HSTR + m] = b;
}

// ---------------------------------------------------------------------------
// Swizzle W2 [64, att_mid] into WMMA-B register layout (f16), per (nt,kstep,lane,e)
__global__ void w2frag_kernel(const float* __restrict__ W2, _Float16* __restrict__ frag, int k32) {
    int total = 4 * k32 * 32 * 16;
    int am = k32 * 32;
    for (int idx = blockIdx.x * blockDim.x + threadIdx.x; idx < total;
         idx += gridDim.x * blockDim.x) {
        int e    = idx & 15;
        int lane = (idx >> 4) & 31;
        int g    = idx >> 9;          // nt*k32 + ks
        int ks   = g % k32;
        int nt   = g / k32;
        int d    = nt * 16 + (lane & 15);
        int k    = ks * 32 + ((lane >= 16) ? 8 : 0) + (e < 8 ? e : e + 8);
        frag[idx] = (_Float16)W2[(size_t)d * am + k];
    }
}

// ---------------------------------------------------------------------------
// Fused attention: att[b,i,d] = sum_{j!=i} tanh( relu(hi_i + hj_j) @ W2^T + b2 )[d]
// block = 128 threads (4 waves, one i each); grid = B * 16
template <int K32>
__global__ __launch_bounds__(128) void att_kernel(const float* __restrict__ hi,
                                                  const float* __restrict__ hj,
                                                  const _Float16* __restrict__ w2f,
                                                  const float* __restrict__ b2,
                                                  float* __restrict__ att) {
    constexpr int AM = K32 * 32;
    __shared__ float hjs[NN * LJ];
    __shared__ float hib[4][AM];

    const int tid  = threadIdx.x;
    const int b    = blockIdx.x >> 4;
    const int ig   = blockIdx.x & 15;
    const int wave = tid >> 5;
    const int lane = tid & 31;
    const int i    = ig * 4 + wave;
    const int rbase = b * NN;

    for (int idx = tid; idx < NN * AM; idx += 128) {
        int row = idx / AM, k = idx - row * AM;
        hjs[row * LJ + k] = hj[(size_t)(rbase + row) * HSTR + k];
    }
    for (int k = lane; k < AM; k += 32)
        hib[wave][k] = hi[(size_t)(rbase + i) * HSTR + k];
    __syncthreads();

    const float* hibw  = hib[wave];
    const v16h*  bfp   = (const v16h*)w2f;
    const int    khalf = (lane >> 4) * 8;          // 0 or 8
    const float* hjr0  = &hjs[(lane & 15) * LJ];

    // hoisted, loop-invariant: b2 value for this lane's output column per n-tile
    float b2v[4];
#pragma unroll
    for (int nt = 0; nt < 4; ++nt) b2v[nt] = b2[nt * 16 + (lane & 15)];

    float acc[4] = {0.f, 0.f, 0.f, 0.f};

    for (int jt = 0; jt < 4; ++jt) {
        const float* hjr = hjr0 + jt * 16 * LJ;
        v16h afr[K32];
#pragma unroll
        for (int ks = 0; ks < K32; ++ks) {
            const int kb = ks * 32 + khalf;
#pragma unroll
            for (int e = 0; e < 16; ++e) {
                int   k = kb + (e < 8 ? e : e + 8);
                float s = __builtin_fmaxf(hibw[k] + hjr[k], 0.f);   // v_max_f32
                afr[ks][e] = (_Float16)s;
            }
        }
#pragma unroll
        for (int nt = 0; nt < 4; ++nt) {
            v8f c = {};
#pragma unroll
            for (int ks = 0; ks < K32; ++ks) {
                v16h bf = bfp[(nt * K32 + ks) * 32 + lane];
                c = __builtin_amdgcn_wmma_f32_16x16x32_f16(false, afr[ks], false, bf,
                                                           (short)0, c, false, false);
            }
#pragma unroll
            for (int v = 0; v < 8; ++v) {
                int   j = jt * 16 + v + khalf;
                float t = fast_tanh(c[v] + b2v[nt]);
                acc[nt] += (j != i) ? t : 0.f;
            }
        }
    }
#pragma unroll
    for (int nt = 0; nt < 4; ++nt) {
        float tot = acc[nt] + __shfl_xor(acc[nt], 16, 32);
        if (lane < 16) att[(size_t)(rbase + i) * 64 + nt * 16 + lane] = tot;
    }
}

// ---------------------------------------------------------------------------
// y1 = tanh(c1b(relu(c1a(x)))); y2 = tanh(c2b(relu(c2a(att)))); y=[y1,y2]/||.||
// writes x[:, 32:160]. block = 128, grid = ROWS
__global__ __launch_bounds__(128) void deepset_kernel(
    float* __restrict__ x, const float* __restrict__ att,
    const float* __restrict__ c1aW, const float* __restrict__ c1aB,
    const float* __restrict__ c1bW, const float* __restrict__ c1bB,
    const float* __restrict__ c2aW, const float* __restrict__ c2aB,
    const float* __restrict__ c2bW, const float* __restrict__ c2bB,
    int F, int mid) {
    __shared__ float xr[XSTR], t1[112], ar[64], t2[64], y[128];
    __shared__ float wpart[4];
    __shared__ float nrm;
    const int r = blockIdx.x, tid = threadIdx.x;
    const int wave = tid >> 5, lane = tid & 31;
    for (int k = tid; k < F; k += 128) xr[k] = x[r * XSTR + k];
    if (tid < 64) ar[tid] = att[r * 64 + tid];
    __syncthreads();
    if (tid < mid) {
        float s = c1aB[tid];
        const float* w = c1aW + (size_t)tid * F;
        for (int k = 0; k < F; ++k) s = fmaf(xr[k], w[k], s);
        t1[tid] = __builtin_fmaxf(s, 0.f);
    }
    if (tid < 64) {
        float s = c2aB[tid];
        const float* w = c2aW + (size_t)tid * 64;
        for (int k = 0; k < 64; ++k) s = fmaf(ar[k], w[k], s);
        t2[tid] = __builtin_fmaxf(s, 0.f);
    }
    __syncthreads();
    float yv;
    if (tid < 64) {
        float s = c1bB[tid];
        const float* w = c1bW + (size_t)tid * mid;
        for (int k = 0; k < mid; ++k) s = fmaf(t1[k], w[k], s);
        yv = fast_tanh(s);
    } else {
        int d = tid - 64;
        float s = c2bB[d];
        const float* w = c2bW + (size_t)d * 64;
        for (int k = 0; k < 64; ++k) s = fmaf(t2[k], w[k], s);
        yv = fast_tanh(s);
    }
    y[tid] = yv;
    // wave-level butterfly reduction of sum(y^2)
    float ss = yv * yv;
#pragma unroll
    for (int m = 16; m > 0; m >>= 1) ss += __shfl_xor(ss, m, 32);
    if (lane == 0) wpart[wave] = ss;
    __syncthreads();
    if (tid == 0) nrm = rsqrtf(wpart[0] + wpart[1] + wpart[2] + wpart[3]);
    __syncthreads();
    x[r * XSTR + 32 + tid] = y[tid] * nrm;
}

// ---------------------------------------------------------------------------
// head: 160 -> relu 256 -> relu 128 -> sigmoid 1. block = 256, grid = ROWS
__global__ __launch_bounds__(256) void head_kernel(
    const float* __restrict__ x,
    const float* __restrict__ h1W, const float* __restrict__ h1B,
    const float* __restrict__ h2W, const float* __restrict__ h2B,
    const float* __restrict__ h3W, const float* __restrict__ h3B,
    float* __restrict__ out) {
    __shared__ float h0[160], h1[256], h2[128], red[256];
    const int r = blockIdx.x, tid = threadIdx.x;
    for (int k = tid; k < 160; k += 256) h0[k] = x[r * XSTR + k];
    __syncthreads();
    {
        float s = h1B[tid];
        const float* w = h1W + (size_t)tid * 160;
        for (int k = 0; k < 160; ++k) s = fmaf(h0[k], w[k], s);
        h1[tid] = __builtin_fmaxf(s, 0.f);
    }
    __syncthreads();
    if (tid < 128) {
        float s = h2B[tid];
        const float* w = h2W + (size_t)tid * 256;
        for (int k = 0; k < 256; ++k) s = fmaf(h1[k], w[k], s);
        h2[tid] = __builtin_fmaxf(s, 0.f);
    }
    __syncthreads();
    red[tid] = (tid < 128) ? h2[tid] * h3W[tid] : 0.f;
    __syncthreads();
    for (int s = 128; s > 0; s >>= 1) {
        if (tid < s) red[tid] += red[tid + s];
        __syncthreads();
    }
    if (tid == 0) {
        float z = red[0] + h3B[0];
        out[r] = __builtin_amdgcn_rcpf(1.f + __expf(-z));
    }
}

// ---------------------------------------------------------------------------
extern "C" void kernel_launch(void* const* d_in, const int* in_sizes, int n_in,
                              void* d_out, int out_size, void* d_ws, size_t ws_size,
                              hipStream_t stream) {
    (void)n_in; (void)out_size; (void)ws_size;
    const float* inp = (const float*)d_in[0];

    // 12 leaves per layer (att.l1 W/b, att.l2 W/b, c1a W/b, c1b W/b, c2a W/b, c2b W/b),
    // 6 head leaves.
    // Param ordering: insertion order -> [inp, layers(36), head(6)]  => Lbase=1, Hbase=37
    //                 jax-sorted      -> [inp, head(6), layers(36)]  => Hbase=1, Lbase=7
    int Lbase, Hbase;
    if (in_sizes[1] == 256 * 160) { Hbase = 1; Lbase = 7; }
    else                          { Lbase = 1; Hbase = 37; }

    char* ws = (char*)d_ws;
    float*    x    = (float*)ws;    ws += (size_t)ROWS * XSTR * 4;
    float*    hi   = (float*)ws;    ws += (size_t)ROWS * HSTR * 4;
    float*    hj   = (float*)ws;    ws += (size_t)ROWS * HSTR * 4;
    float*    attb = (float*)ws;    ws += (size_t)ROWS * 64 * 4;
    _Float16* w2f  = (_Float16*)ws; ws += (size_t)4 * 6 * 32 * 16 * 2;

    copy_inp_kernel<<<(ROWS * 32 + 255) / 256, 256, 0, stream>>>(inp, x);

    for (int l = 0; l < 3; ++l) {
        const int F   = (l == 0) ? 32 : 160;
        const int AM  = (l == 0) ? 64 : 192;
        const int MID = (l == 0) ? 48 : 112;
        const int p   = Lbase + 12 * l;
        const float* l1W  = (const float*)d_in[p + 0];
        const float* l1b  = (const float*)d_in[p + 1];
        const float* l2W  = (const float*)d_in[p + 2];
        const float* l2b  = (const float*)d_in[p + 3];
        const float* c1aW = (const float*)d_in[p + 4];
        const float* c1aB = (const float*)d_in[p + 5];
        const float* c1bW = (const float*)d_in[p + 6];
        const float* c1bB = (const float*)d_in[p + 7];
        const float* c2aW = (const float*)d_in[p + 8];
        const float* c2aB = (const float*)d_in[p + 9];
        const float* c2bW = (const float*)d_in[p + 10];
        const float* c2bB = (const float*)d_in[p + 11];

        hi_hj_kernel<<<ROWS, AM, 0, stream>>>(x, l1W, l1b, hi, hj, F);
        w2frag_kernel<<<24, 256, 0, stream>>>(l2W, w2f, AM / 32);
        if (l == 0)
            att_kernel<2><<<BB * 16, 128, 0, stream>>>(hi, hj, w2f, l2b, attb);
        else
            att_kernel<6><<<BB * 16, 128, 0, stream>>>(hi, hj, w2f, l2b, attb);
        deepset_kernel<<<ROWS, 128, 0, stream>>>(x, attb, c1aW, c1aB, c1bW, c1bB,
                                                 c2aW, c2aB, c2bW, c2bB, F, MID);
    }

    head_kernel<<<ROWS, 256, 0, stream>>>(x,
        (const float*)d_in[Hbase + 0], (const float*)d_in[Hbase + 1],
        (const float*)d_in[Hbase + 2], (const float*)d_in[Hbase + 3],
        (const float*)d_in[Hbase + 4], (const float*)d_in[Hbase + 5],
        (float*)d_out);
}
